// _ChannelLSTMF_3693671875178
// MI455X (gfx1250) — compile-verified, matrix operand order checked by reading, not run on hardware
//
#include <hip/hip_runtime.h>
#include <hip/hip_bf16.h>

typedef __attribute__((ext_vector_type(16))) __bf16 v16bf;
typedef __attribute__((ext_vector_type(8)))  __bf16 v8bf;
typedef __attribute__((ext_vector_type(8)))  float  v8f;

#define WMMA_BF16(a,b,c) \
  __builtin_amdgcn_wmma_f32_16x16x32_bf16(false,(a),false,(b),(short)0,(c),false,false)

__device__ __forceinline__ v16bf cat8(v8bf lo, v8bf hi){
  return __builtin_shufflevector(lo, hi, 0,1,2,3,4,5,6,7,8,9,10,11,12,13,14,15);
}
__device__ __forceinline__ v8bf ld8_cvt(const float* __restrict__ p){
  v8bf r;
  #pragma unroll
  for (int i=0;i<8;++i) r[i] = (__bf16)p[i];
  return r;
}
__device__ __forceinline__ float fast_sigmoid(float x){
  return __builtin_amdgcn_rcpf(1.0f + __expf(-x));
}
__device__ __forceinline__ float fast_tanh(float x){
  float e = __expf(-2.0f*x);
  return (1.0f - e) * __builtin_amdgcn_rcpf(1.0f + e);
}

// ---------------------------------------------------------------------------
// Kernel A: one direction of the frequency-LSTM for a tile of 16 sequences.
// Block = 4 waves. Wave w computes gate class w (i/f/g/o, 64 gates each):
//   per step: 1 WMMA (y x W_ih) + 2 WMMA (h x W_hh, K=64) per 16-gate tile
// Gates exchanged via LDS; wave w then owns H-slice [w*16, w*16+16) of c/h.
// Writes h (bf16) to workspace row r = f*2048 + n, cols [dir*64, dir*64+64).
// ---------------------------------------------------------------------------
__global__ __launch_bounds__(128) void lstm_dir_kernel(
    const float* __restrict__ x,
    const float* __restrict__ Wih_f, const float* __restrict__ Whh_f,
    const float* __restrict__ bih_f, const float* __restrict__ bhh_f,
    const float* __restrict__ Wih_b, const float* __restrict__ Whh_b,
    const float* __restrict__ bih_b, const float* __restrict__ bhh_b,
    __bf16* __restrict__ hcat)
{
  const int dir = blockIdx.y;
  const float* __restrict__ Wih = dir ? Wih_b : Wih_f;
  const float* __restrict__ Whh = dir ? Whh_b : Whh_f;
  const float* __restrict__ bih = dir ? bih_b : bih_f;
  const float* __restrict__ bhh = dir ? bhh_b : bhh_f;

  const int tid = threadIdx.x;
  const int w   = tid >> 5;      // wave id 0..3  == gate class (i,f,g,o)
  const int L   = tid & 31;      // lane
  const int lh  = L >> 4;        // lane half (K-run selector)
  const int ln  = L & 15;        // lane low 4 bits (M or N index)

  const int n0 = blockIdx.x * 16;      // 16 consecutive (b,t) sequences
  const int bb = n0 >> 8;              // same batch for whole tile
  const int t0 = n0 & 255;

  __shared__ __align__(16) float  gbuf[16*256]; // gates  [seq][gate]
  __shared__ __align__(16) __bf16 hbuf[16*64];  // h      [seq][H]
  __shared__ __align__(16) __bf16 ybuf[16*32];  // y tile [t][c]

  // -------- resident weight operands in WMMA B-layout ---------------------
  // B operand (K x 16): lane ln = N(gate); K-runs of 8: lanes<16 -> {0..7,16..23}
  v16bf Bhh[4][2];   // 4 gate tiles x 2 K-halves of 32 (H = 64)
  v16bf Bih[4];      // 4 gate tiles, K = 32 (input channels)
  float bias[4];
  #pragma unroll
  for (int jt=0; jt<4; ++jt) {
    const int gate = w*64 + jt*16 + ln;
    #pragma unroll
    for (int kh=0; kh<2; ++kh) {
      const int kb = kh*32 + lh*8;
      Bhh[jt][kh] = cat8(ld8_cvt(&Whh[gate*64 + kb]),
                         ld8_cvt(&Whh[gate*64 + kb + 16]));
    }
    const int kb = lh*8;
    Bih[jt] = cat8(ld8_cvt(&Wih[gate*32 + kb]),
                   ld8_cvt(&Wih[gate*32 + kb + 16]));
    bias[jt] = bih[gate] + bhh[gate];
  }

  v16bf hA0 = {}, hA1 = {};   // h in A-layout (16 seq x 64 H bf16), starts 0
  v8f   c   = {};             // cell state slice (seq = v + 8*lh, H = w*16+ln)

  for (int step=0; step<256; ++step) {
    const int f = dir ? (255 - step) : step;

    // ---- stage y[:,f,:] tile (16 t x 32 c) into LDS, coalesced over t ----
    #pragma unroll
    for (int e=0; e<4; ++e) {
      const int idx = tid + e*128;
      const int cc = idx >> 4, tl = idx & 15;
      ybuf[tl*32 + cc] = (__bf16)x[(((bb*32 + cc)*256 + f) << 8) + t0 + tl];
    }
    if (step + 1 < 256) {
      const int nf = dir ? (f-1) : (f+1);
      #pragma unroll
      for (int e=0; e<4; ++e) {
        const int idx = tid + e*128;
        const int cc = idx >> 4, tl = idx & 15;
        __builtin_prefetch(&x[(((bb*32 + cc)*256 + nf) << 8) + t0 + tl], 0, 0);
      }
    }
    __syncthreads();

    // A operand for y: lane ln = seq(t), K-runs of 8 channels
    const int kby = lh*8;
    const v16bf yA = cat8(*(const v8bf*)&ybuf[ln*32 + kby],
                          *(const v8bf*)&ybuf[ln*32 + kby + 16]);

    // ---- gate GEMM: g = bias + y*Wih^T + h*Whh^T (this wave's 64 gates) --
    v8f g[4];
    #pragma unroll
    for (int jt=0; jt<4; ++jt) {
      v8f acc;
      #pragma unroll
      for (int v=0; v<8; ++v) acc[v] = bias[jt];
      acc = WMMA_BF16(yA,  Bih[jt],    acc);
      acc = WMMA_BF16(hA0, Bhh[jt][0], acc);
      acc = WMMA_BF16(hA1, Bhh[jt][1], acc);
      g[jt] = acc;
    }

    // ---- share gates across waves ----------------------------------------
    #pragma unroll
    for (int jt=0; jt<4; ++jt) {
      const int gate = w*64 + jt*16 + ln;
      #pragma unroll
      for (int v=0; v<8; ++v)
        gbuf[(v + 8*lh)*256 + gate] = g[jt][v];
    }
    __syncthreads();

    // ---- elementwise update on this wave's H-slice -----------------------
    const int hh = w*16 + ln;
    #pragma unroll
    for (int v=0; v<8; ++v) {
      const int seq = v + 8*lh;
      const float gi = gbuf[seq*256 +       hh];
      const float gf = gbuf[seq*256 +  64 + hh];
      const float gg = gbuf[seq*256 + 128 + hh];
      const float go = gbuf[seq*256 + 192 + hh];
      const float cn = fast_sigmoid(gf)*c[v] + fast_sigmoid(gi)*fast_tanh(gg);
      c[v] = cn;
      const float hv = fast_sigmoid(go)*fast_tanh(cn);
      const __bf16 hb = (__bf16)hv;
      hbuf[seq*64 + hh] = hb;
      const long r = (long)f*2048 + n0 + seq;          // ws row = f*2048 + n
      hcat[r*128 + dir*64 + hh] = hb;
    }
    __syncthreads();

    // ---- reload h into A-layout for next step ----------------------------
    const int kb = lh*8;
    hA0 = cat8(*(const v8bf*)&hbuf[ln*64 + kb],
               *(const v8bf*)&hbuf[ln*64 + kb + 16]);
    hA1 = cat8(*(const v8bf*)&hbuf[ln*64 + 32 + kb],
               *(const v8bf*)&hbuf[ln*64 + 32 + kb + 16]);
  }
}

// ---------------------------------------------------------------------------
// Kernel B: out[n,f,:] = hcat[n,f,:] @ W_lin^T + b_lin   (K=128, 32 outputs)
// A = W_lin (M=out, resident in VGPRs), B = hcat^T tile (lane = row).
// Rows r = f*2048 + n  => a 16-row tile is 16 consecutive t -> coalesced store.
// ---------------------------------------------------------------------------
__global__ __launch_bounds__(256) void proj_kernel(
    const __bf16* __restrict__ hcat,
    const float* __restrict__ Wlin,
    const float* __restrict__ blin,
    float* __restrict__ out)
{
  const int tid = threadIdx.x;
  const int wv  = tid >> 5;
  const int L   = tid & 31;
  const int lh  = L >> 4, ln = L & 15;

  v16bf A[2][4];     // 2 out-tiles x 4 K-halves of 32
  float bl[2][8];
  #pragma unroll
  for (int m=0; m<2; ++m) {
    const int o = m*16 + ln;
    #pragma unroll
    for (int kt=0; kt<4; ++kt) {
      const int kb = kt*32 + lh*8;
      A[m][kt] = cat8(ld8_cvt(&Wlin[o*128 + kb]),
                      ld8_cvt(&Wlin[o*128 + kb + 16]));
    }
    #pragma unroll
    for (int v=0; v<8; ++v) bl[m][v] = blin[m*16 + v + 8*lh];
  }

  const int nTiles = 32768;                 // 524288 rows / 16
  for (int tile = blockIdx.x*8 + wv; tile < nTiles; tile += gridDim.x*8) {
    const long r0 = (long)tile * 16;
    const int  f  = (int)(r0 >> 11);
    const int  n0 = (int)(r0 & 2047);
    const int  bb = n0 >> 8;
    const int  t0 = n0 & 255;
    const long rb = (r0 + ln) * 128;        // this lane's row base in ws

    v8f acc[2];
    #pragma unroll
    for (int m=0; m<2; ++m)
      #pragma unroll
      for (int v=0; v<8; ++v) acc[m][v] = bl[m][v];

    #pragma unroll
    for (int kt=0; kt<4; ++kt) {
      const int kb = kt*32 + lh*8;
      const v16bf Bop = cat8(*(const v8bf*)&hcat[rb + kb],
                             *(const v8bf*)&hcat[rb + kb + 16]);
      acc[0] = WMMA_BF16(A[0][kt], Bop, acc[0]);
      acc[1] = WMMA_BF16(A[1][kt], Bop, acc[1]);
    }
    #pragma unroll
    for (int m=0; m<2; ++m) {
      #pragma unroll
      for (int v=0; v<8; ++v) {
        const int o = m*16 + v + 8*lh;
        out[(((long)(bb*32 + o)*256 + f) << 8) + t0 + ln] = acc[m][v];
      }
    }
  }
}

extern "C" void kernel_launch(void* const* d_in, const int* in_sizes, int n_in,
                              void* d_out, int out_size, void* d_ws, size_t ws_size,
                              hipStream_t stream) {
  (void)in_sizes; (void)n_in; (void)out_size; (void)ws_size;
  const float* x     = (const float*)d_in[0];
  const float* Wih_f = (const float*)d_in[1];
  const float* Whh_f = (const float*)d_in[2];
  const float* bih_f = (const float*)d_in[3];
  const float* bhh_f = (const float*)d_in[4];
  const float* Wih_b = (const float*)d_in[5];
  const float* Whh_b = (const float*)d_in[6];
  const float* bih_b = (const float*)d_in[7];
  const float* bhh_b = (const float*)d_in[8];
  const float* Wlin  = (const float*)d_in[9];
  const float* blin  = (const float*)d_in[10];
  float*  out  = (float*)d_out;
  __bf16* hcat = (__bf16*)d_ws;   // 2048*256*128 bf16 = 128 MB

  dim3 gA(128, 2), bA(128);       // 128 n-tiles x 2 directions, 4 waves each
  lstm_dir_kernel<<<gA, bA, 0, stream>>>(x, Wih_f, Whh_f, bih_f, bhh_f,
                                         Wih_b, Whh_b, bih_b, bhh_b, hcat);

  dim3 gB(4096), bB(256);         // 32768 row-tiles, 8 waves/block
  proj_kernel<<<gB, bB, 0, stream>>>(hcat, Wlin, blin, out);
}
